// Modelev12_64690797412523
// MI455X (gfx1250) — matrix operand order repaired and checked by reading, not compiled, not run on hardware
//
#include <hip/hip_runtime.h>

// ---------------------------------------------------------------------------
// 2-layer LSTM (H=48) recurrence on gfx1250, v_wmma_f32_16x16x32_f16.
// 256 workgroups x 96 threads (3 wave32 waves); each owns 16 batch rows and
// walks all 576 steps with only TWO barriers per step.
//
// Wave w owns hidden columns 16w..16w+15 for ALL four gates; the WMMA C
// layout gives each lane i/f/g/o of the same (m,c) cell -> the LSTM cell
// update runs entirely in registers (c-state in VGPRs).
//
// h-state lives in LDS as f16 in four DOUBLE-BUFFERED 64-col windows
// (h0[0], h0[1], h1[0], h1[1]): same-step reads use parity (t-1)&1, writes
// use parity t&1 -> no WAR barriers.
//
// The output projection is a WMMA tile too: W_lin replicated across all 16
// B-columns over the h1_prev window, so EVERY lane gets out_{t-1} for its 8
// rows in registers -> it is simultaneously the global output (stored by
// wave 0) and the autoregressive input x_t for t >= T (zero LDS, zero
// shuffles). Observed-phase x comes from global (cache-resident).
//
// Activations use the native v_tanh_f32 transcendental when available.
// ---------------------------------------------------------------------------

typedef __attribute__((ext_vector_type(16))) _Float16 v16h;
typedef __attribute__((ext_vector_type(8)))  float    v8f;

#define HH   48
#define TOBS 512
#define TTOT 576
#define HCS  264   // hcat row stride in halves: 4x64-col windows + pad
                   // (528B rows: 16B aligned, 132%64 dwords = 4 -> bank spread)

#if __has_builtin(__builtin_amdgcn_tanhf)
__device__ __forceinline__ float ftanh_(float x) { return __builtin_amdgcn_tanhf(x); }
#else
__device__ __forceinline__ float ftanh_(float x) {
    float e = __builtin_amdgcn_exp2f(x * -2.885390081777927f);  // exp(-2x)
    return (1.f - e) * __builtin_amdgcn_rcpf(1.f + e);
}
#endif
__device__ __forceinline__ float fsig_(float x) { return 0.5f * ftanh_(0.5f * x) + 0.5f; }

__device__ __forceinline__ v8f wmma_f16(v16h a, v16h b, v8f c) {
    // (neg_a, A, neg_b, B, c_mod, C, reuse_a, reuse_b)
    return __builtin_amdgcn_wmma_f32_16x16x32_f16(false, a, false, b, (short)0, c,
                                                  false, false);
}

__global__ __launch_bounds__(96, 1) void lstm_wmma_persistent(
    const float* __restrict__ xg,
    const float* __restrict__ Wih0, const float* __restrict__ Whh0,
    const float* __restrict__ bih0, const float* __restrict__ bhh0,
    const float* __restrict__ Wih1, const float* __restrict__ Whh1,
    const float* __restrict__ bih1, const float* __restrict__ bhh1,
    const float* __restrict__ Wlin, const float* __restrict__ blin,
    float* __restrict__ outg) {

    // windows: h0[p] at 64*p, h1[p] at 128 + 64*p  (cols 48-63 of each stay 0)
    __shared__ _Float16 hcat[16 * HCS];

    const int tid  = threadIdx.x;
    const int lane = tid & 31;
    const int wv   = tid >> 5;               // 0..2: hidden column group
    const int row0 = blockIdx.x * 16;

    for (int i = tid; i < 16 * HCS; i += 96) hcat[i] = (_Float16)0.f;

    // Per-lane hidden column and loop-invariant scalars.
    const int   hc     = 16 * wv + (lane & 15);   // hidden col 0..47
    const float blin_v = blin[0];
    float b0c[4], b1c[4], w0x[4];
#pragma unroll
    for (int g = 0; g < 4; ++g) {
        const int n = g * HH + hc;           // gate-matrix row index
        b0c[g] = bih0[n] + bhh0[n];
        b1c[g] = bih1[n] + bhh1[n];
        w0x[g] = Wih0[n];                    // W_ih0 is (192,1)
    }

    // Weight B-fragments (16-bit 32x16 layout: lane&15 = N column,
    // K = 8*(lane>>4) + (j&7) + 16*(j>>3) within the 32-wide K chunk).
    const int khalf = (lane >> 4) << 4;
    v16h w0f[4][2];   // layer0: Whh0^T   over h0 window (rows>=48 zero)
    v16h w1f[4][4];   // layer1: chunks 0,1 = Wih1^T over h0 window,
                      //         chunks 2,3 = Whh1^T over h1 window (rows>=48 zero)
    v16h wof[2];      // out:    W_lin replicated in all 16 cols over h1 window
#pragma unroll
    for (int g = 0; g < 4; ++g) {
        const int n = g * HH + hc;
#pragma unroll
        for (int ch = 0; ch < 2; ++ch) {
#pragma unroll
            for (int j = 0; j < 16; ++j) {
                const int k = ch * 32 + ((j & 8) << 1) + ((lane >> 4) << 3) + (j & 7);
                w0f[g][ch][j]     = (_Float16)((k < HH) ? Whh0[n * HH + k] : 0.f);
                w1f[g][ch][j]     = (_Float16)((k < HH) ? Wih1[n * HH + k] : 0.f);
                w1f[g][ch + 2][j] = (_Float16)((k < HH) ? Whh1[n * HH + k] : 0.f);
            }
        }
    }
#pragma unroll
    for (int ch = 0; ch < 2; ++ch) {
#pragma unroll
        for (int j = 0; j < 16; ++j) {
            const int k = ch * 32 + ((j & 8) << 1) + ((lane >> 4) << 3) + (j & 7);
            wof[ch][j] = (_Float16)((k < HH) ? Wlin[k] : 0.f);  // same for every column
        }
    }

    // A-fragment addressing (16-bit 16x32 layout): lane&15 = row M,
    // kg = 8*(lane>>4); halves 0-7 -> K=kg.., halves 8-15 -> K=16+kg..
    const int am = lane & 15;
    const int kg = (lane >> 4) << 3;
    const int mh = 8 * (lane >> 4);          // C-tile row offset for this lane

    float c0r[8] = {0.f};                    // cell state (m = mh+r, col hc)
    float c1r[8] = {0.f};

    for (int t = 0; t < TTOT; ++t) {
        const int pp = 64 * ((t + 1) & 1);   // prev-parity window offset
        const int cp = 64 * (t & 1);         // current-parity window offset

        __syncthreads();  // B_top: h0/h1 of step t-1 visible

        // ---- A fragments: h0_prev (a0,a1) and h1_prev (a2,a3) -----------
        v16h a0, a1, a2, a3;
#pragma unroll
        for (int j = 0; j < 8; ++j) {
            a0[j]     = hcat[am * HCS + pp + kg + j];
            a0[8 + j] = hcat[am * HCS + pp + 16 + kg + j];
            a1[j]     = hcat[am * HCS + pp + 32 + kg + j];
            a1[8 + j] = hcat[am * HCS + pp + 48 + kg + j];
            a2[j]     = hcat[am * HCS + 128 + pp + kg + j];
            a2[8 + j] = hcat[am * HCS + 128 + pp + 16 + kg + j];
            a3[j]     = hcat[am * HCS + 128 + pp + 32 + kg + j];
            a3[8 + j] = hcat[am * HCS + 128 + pp + 48 + kg + j];
        }

        float xl[8];
        if (t < TOBS) {
#pragma unroll
            for (int r = 0; r < 8; ++r)
                xl[r] = xg[(size_t)(row0 + mh + r) * TOBS + t];
        }

        // ---- layer 0 gates + output tile (all on prev-parity state) -----
        v8f ai = {}, af = {}, ag = {}, ao = {}, aout = {};
        ai   = wmma_f16(a0, w0f[0][0], ai);  ai   = wmma_f16(a1, w0f[0][1], ai);
        af   = wmma_f16(a0, w0f[1][0], af);  af   = wmma_f16(a1, w0f[1][1], af);
        ag   = wmma_f16(a0, w0f[2][0], ag);  ag   = wmma_f16(a1, w0f[2][1], ag);
        ao   = wmma_f16(a0, w0f[3][0], ao);  ao   = wmma_f16(a1, w0f[3][1], ao);
        aout = wmma_f16(a2, wof[0], aout);   aout = wmma_f16(a3, wof[1], aout);

        // out_{t-1} = aout + blin, identical in every lane (W_lin replicated)
        if (wv == 0 && (lane & 15) == 0 && t > 0) {
#pragma unroll
            for (int r = 0; r < 8; ++r)
                outg[(size_t)(row0 + mh + r) * TTOT + (t - 1)] = aout[r] + blin_v;
        }

        // ---- layer 0 cell update (registers only) -----------------------
#pragma unroll
        for (int r = 0; r < 8; ++r) {
            const float xv = (t < TOBS) ? xl[r] : (aout[r] + blin_v);
            const float cn = fsig_(af[r] + xv * w0x[1] + b0c[1]) * c0r[r]
                           + fsig_(ai[r] + xv * w0x[0] + b0c[0])
                           * ftanh_(ag[r] + xv * w0x[2] + b0c[2]);
            c0r[r] = cn;
            const float h = fsig_(ao[r] + xv * w0x[3] + b0c[3]) * ftanh_(cn);
            hcat[(mh + r) * HCS + cp + hc] = (_Float16)h;
        }
        __syncthreads();  // B_mid: h0_t visible

        // ---- layer 1: [h0_t (fresh) ; h1_{t-1} (a2,a3 reused)] ----------
        v16h b0, b1;
#pragma unroll
        for (int j = 0; j < 8; ++j) {
            b0[j]     = hcat[am * HCS + cp + kg + j];
            b0[8 + j] = hcat[am * HCS + cp + 16 + kg + j];
            b1[j]     = hcat[am * HCS + cp + 32 + kg + j];
            b1[8 + j] = hcat[am * HCS + cp + 48 + kg + j];
        }
        v8f bi = {}, bf = {}, bg = {}, bo = {};
        bi = wmma_f16(b0, w1f[0][0], bi); bi = wmma_f16(b1, w1f[0][1], bi);
        bi = wmma_f16(a2, w1f[0][2], bi); bi = wmma_f16(a3, w1f[0][3], bi);
        bf = wmma_f16(b0, w1f[1][0], bf); bf = wmma_f16(b1, w1f[1][1], bf);
        bf = wmma_f16(a2, w1f[1][2], bf); bf = wmma_f16(a3, w1f[1][3], bf);
        bg = wmma_f16(b0, w1f[2][0], bg); bg = wmma_f16(b1, w1f[2][1], bg);
        bg = wmma_f16(a2, w1f[2][2], bg); bg = wmma_f16(a3, w1f[2][3], bg);
        bo = wmma_f16(b0, w1f[3][0], bo); bo = wmma_f16(b1, w1f[3][1], bo);
        bo = wmma_f16(a2, w1f[3][2], bo); bo = wmma_f16(a3, w1f[3][3], bo);

        // ---- layer 1 cell update ----------------------------------------
#pragma unroll
        for (int r = 0; r < 8; ++r) {
            const float cn = fsig_(bf[r] + b1c[1]) * c1r[r]
                           + fsig_(bi[r] + b1c[0]) * ftanh_(bg[r] + b1c[2]);
            c1r[r] = cn;
            const float h = fsig_(bo[r] + b1c[3]) * ftanh_(cn);
            hcat[(mh + r) * HCS + 128 + cp + hc] = (_Float16)h;
        }
        // next B_top orders h0_t/h1_t against step t+1
    }

    // ---- epilogue: output for the final step ----------------------------
    __syncthreads();
    if (tid < 16) {
        float ov = blin_v;
        const int wbase = 128 + 64 * ((TTOT - 1) & 1);
#pragma unroll
        for (int c = 0; c < HH; ++c)
            ov += (float)hcat[tid * HCS + wbase + c] * Wlin[c];
        outg[(size_t)(row0 + tid) * TTOT + (TTOT - 1)] = ov;
    }
}

extern "C" void kernel_launch(void* const* d_in, const int* in_sizes, int n_in,
                              void* d_out, int out_size, void* d_ws, size_t ws_size,
                              hipStream_t stream) {
    (void)in_sizes; (void)n_in; (void)out_size; (void)d_ws; (void)ws_size;
    const float* x    = (const float*)d_in[0];
    const float* Wih0 = (const float*)d_in[1];
    const float* Whh0 = (const float*)d_in[2];
    const float* bih0 = (const float*)d_in[3];
    const float* bhh0 = (const float*)d_in[4];
    const float* Wih1 = (const float*)d_in[5];
    const float* Whh1 = (const float*)d_in[6];
    const float* bih1 = (const float*)d_in[7];
    const float* bhh1 = (const float*)d_in[8];
    const float* Wlin = (const float*)d_in[9];
    const float* blin = (const float*)d_in[10];
    float* out = (float*)d_out;

    dim3 grid(4096 / 16);   // 256 workgroups, one 16-row batch tile each
    dim3 block(96);         // 3 wave32 waves -> hidden column groups
    lstm_wmma_persistent<<<grid, block, 0, stream>>>(
        x, Wih0, Whh0, bih0, bhh0, Wih1, Whh1, bih1, bhh1, Wlin, blin, out);
}
